// NSFPosteriorFlow_22832046145987
// MI455X (gfx1250) — matrix operand order repaired and checked by reading, not compiled, not run on hardware
//
#include <hip/hip_runtime.h>
#include <hip/hip_bf16.h>

typedef __attribute__((ext_vector_type(16))) __bf16 v16bf;
typedef __attribute__((ext_vector_type(8)))  float  v8f;

#define F_DIM  16
#define HID    256
#define CDIM   256
#define NLAYER 12
#define KBIN   16
#define MOUT   47
#define POUT   752            // F_DIM * MOUT
#define TBND   3.0f
#define MINW   0.001f
#define MIND   0.001f
#define SSCALE 0.0625f        // 1/sqrt(HID)

#define BT     64             // batch rows per block
#define HP     264            // f32 LDS row stride (floats)
#define HBP    264            // bf16 LDS row stride (halves)
#define MAXCHUNK 16384

// ---------------- fragment loaders ----------------
// B fragment: 16 consecutive bf16 per lane (our packed layout).
__device__ __forceinline__ v16bf ldbfrag(const __bf16* p) {
  union { int4 q[2]; v16bf v; } u;
  u.q[0] = *(const int4*)(p);
  u.q[1] = *(const int4*)(p + 8);
  return u.v;
}
// A fragment: halves 0..7 at +0, halves 8..15 at +16 halves (ISA A layout: K+16).
__device__ __forceinline__ v16bf ldafrag(const __bf16* p) {
  union { int4 q[2]; v16bf v; } u;
  u.q[0] = *(const int4*)(p);
  u.q[1] = *(const int4*)(p + 16);
  return u.v;
}

// Load 8 A-fragments (K=256) for this wave's 16-row M tile from a bf16
// row-major buffer (global ctx rows or LDS activation rows).
__device__ __forceinline__ void load_a8(v16bf a[8], const __bf16* base,
                                        int stride, int mbase, int lane) {
  int m    = mbase + (lane & 15);
  int koff = (lane >> 4) << 3;      // lanes>=16 start at K+8
  const __bf16* row = base + (size_t)m * stride + koff;
#pragma unroll
  for (int kc = 0; kc < 8; ++kc) a[kc] = ldafrag(row + kc * 32);
}

// ---------------- generic WMMA stage (fully specialized by template) ----
// out[m, n] = bias0[n] (+bias1[n]) (+hadd[m,n]) + sum_k A[m,k] * Bfrag[k,n]
template<int NT, bool HASB1, bool HASADD, bool WH, bool WHB, bool WP, int KC2>
__device__ __forceinline__ void run_stage(
    const v16bf a[8], const __bf16* bfrag,
    const float* bias0, const float* bias1,
    const float* haddsm,          // LDS f32 residual input (HASADD)
    float* hout,                  // LDS f32 output (WH)
    __bf16* hbout,                // LDS bf16 relu(out) (WHB)
    float* pout,                  // global f32 output rows (WP), stride POUT
    const v16bf* a2, const __bf16* bfrag2,   // optional extra product (KC2)
    int mbase, int nhalf, int lane) {
  const int nlo  = lane & 15;
  const int moff = (lane >> 4) << 3;    // C/D layout: lanes>=16 are rows M+8

  auto tile = [&](int nt) {
    const int n = nt * 16 + nlo;
    const __bf16* fp = bfrag + ((size_t)nt * 8) * 512 + lane * 16;
    __builtin_prefetch(fp + (size_t)2 * 8 * 512, 0, 1);   // next tile for this wave
    // preload ALL B fragments for this tile -> one clause, one wait,
    // then the WMMA chain issues back-to-back
    v16bf bm[8];
#pragma unroll
    for (int kc = 0; kc < 8; ++kc) bm[kc] = ldbfrag(fp + (size_t)kc * 512);
    v16bf bx;
    if (KC2) bx = ldbfrag(bfrag2 + ((size_t)nt * KC2) * 512 + lane * 16);

    float binit = bias0[n];
    if (HASB1) binit += bias1[n];
    v8f acc;
#pragma unroll
    for (int r = 0; r < 8; ++r) {
      float v = binit;
      if (HASADD) v += haddsm[(mbase + moff + r) * HP + n];
      acc[r] = v;
    }
#pragma unroll
    for (int kc = 0; kc < 8; ++kc)
      acc = __builtin_amdgcn_wmma_f32_16x16x32_bf16(false, a[kc], false, bm[kc],
                                                    (short)0, acc, false, false);
    if (KC2)
      acc = __builtin_amdgcn_wmma_f32_16x16x32_bf16(false, a2[0], false, bx,
                                                    (short)0, acc, false, false);
#pragma unroll
    for (int r = 0; r < 8; ++r) {
      int m = mbase + moff + r;
      float v = acc[r];
      if (WH)  hout[m * HP + n] = v;
      if (WHB) hbout[m * HBP + n] = (__bf16)fmaxf(v, 0.f);
      if (WP)  __builtin_nontemporal_store(v, pout + (size_t)m * POUT + n);
    }
  };

  if (NT == 16) {
    // uniform 8 tiles/wave: full unroll lets the scheduler pipeline
    // tile i+1's fragment loads under tile i's WMMA chain
#pragma unroll
    for (int i = 0; i < 8; ++i) tile(nhalf + 2 * i);
  } else {
    for (int nt = nhalf; nt < NT; nt += 2) tile(nt);
  }
}

// ---------------- fused MADE (all matmul stages of one layer) ----------------
__global__ void __launch_bounds__(256) made_kernel(
    const float* __restrict__ Xsrc, const __bf16* __restrict__ ctxb,
    const __bf16* __restrict__ w0f, const __bf16* __restrict__ wcf,
    const __bf16* __restrict__ wrf, const __bf16* __restrict__ wff,
    const float* __restrict__ b0, const float* __restrict__ bc,
    const float* __restrict__ br, const float* __restrict__ bfb,
    float* __restrict__ P, int layer, int rowbase) {
  extern __shared__ char smem[];
  float*  hsm = (float*)smem;                                   // [BT][HP] f32
  __bf16* hb0 = (__bf16*)(smem + (size_t)BT * HP * sizeof(float));
  __bf16* hb1 = hb0 + (size_t)BT * HBP;

  const int tid = threadIdx.x, lane = tid & 31, wave = tid >> 5;
  const int mbase = (wave & 3) * 16, nhalf = wave >> 2;
  const int blockrow = rowbase + blockIdx.x * BT;

  const __bf16* w0 = w0f + (size_t)layer * (16 * 1 * 512);
  const __bf16* wc = wcf + (size_t)layer * (16 * 8 * 512);
  const __bf16* wr = wrf + (size_t)layer * 4 * (16 * 8 * 512);
  const __bf16* wf = wff + (size_t)layer * (47 * 8 * 512);
  const float* b0l = b0 + layer * HID;
  const float* bcl = bc + layer * HID;
  const float* brl = br + layer * 4 * HID;
  const float* bfl = bfb + (size_t)layer * POUT;

  v16bf a[8], ax;
  // ctx A-fragments straight from global bf16 rows
  load_a8(a, ctxb + (size_t)blockrow * CDIM, CDIM, mbase, lane);
  // reversed-x A-fragment, real K = 16 (halves 8..15 map to K>=16 -> zero)
  {
    int m = blockrow + mbase + (lane & 15);
    const float* xr = Xsrc + (size_t)m * F_DIM;
    int kofs = (lane >> 4) << 3;
#pragma unroll
    for (int hh = 0; hh < 8; ++hh) ax[hh] = (__bf16)xr[15 - (kofs + hh)];
#pragma unroll
    for (int hh = 8; hh < 16; ++hh) ax[hh] = (__bf16)0.f;
  }
  // h = ctx@WcT + x_rev@(W0*m0)T + b0 + bc ; keep f32 and relu-bf16 copies
  run_stage<16, true, false, true, true, false, 1>(
      a, wc, b0l, bcl, nullptr, hsm, hb0, nullptr, &ax, w0, mbase, nhalf, lane);
  __syncthreads();
#pragma unroll 1
  for (int blk = 0; blk < 2; ++blk) {
    load_a8(a, hb0, HBP, mbase, lane);   // relu(h)
    run_stage<16, false, false, false, true, false, 0>(
        a, wr + (size_t)(blk * 2 + 0) * (16 * 8 * 512),
        brl + (blk * 2 + 0) * HID, nullptr, nullptr, nullptr, hb1, nullptr,
        nullptr, nullptr, mbase, nhalf, lane);       // t -> relu(t) bf16
    __syncthreads();
    load_a8(a, hb1, HBP, mbase, lane);
    run_stage<16, false, true, true, true, false, 0>(
        a, wr + (size_t)(blk * 2 + 1) * (16 * 8 * 512),
        brl + (blk * 2 + 1) * HID, nullptr, hsm, hsm, hb0, nullptr,
        nullptr, nullptr, mbase, nhalf, lane);       // h += t2 ; refresh relu copy
    __syncthreads();
  }
  load_a8(a, hb0, HBP, mbase, lane);
  run_stage<47, false, false, false, false, true, 0>(
      a, wf, bfl, nullptr, nullptr, nullptr, nullptr,
      P + (size_t)blockIdx.x * BT * POUT, nullptr, nullptr,
      mbase, nhalf, lane);               // p -> global workspace
}

// ---------------- rational-quadratic spline ----------------
__device__ __forceinline__ float softplusf(float v) {
  return (v > 20.f) ? v : log1pf(__expf(v));
}

__global__ void __launch_bounds__(256) spline_kernel(
    const float* __restrict__ Xsrc, const float* __restrict__ P,
    float* __restrict__ Xdst, float* __restrict__ ldout,
    int rowbase, int rows) {
  int e = blockIdx.x * blockDim.x + threadIdx.x;
  if (e >= rows * F_DIM) return;
  int bl = e >> 4, f = e & 15;
  int b = rowbase + bl;
  float x0 = Xsrc[(size_t)b * F_DIM + (15 - f)];     // reversed input
  const float* p = P + (size_t)bl * POUT + f * MOUT;
  bool inside = (x0 >= -TBND) && (x0 <= TBND);
  float xc = fminf(fmaxf(x0, -TBND), TBND);

  float uw[KBIN], uh[KBIN];
#pragma unroll
  for (int i = 0; i < KBIN; ++i) { uw[i] = p[i] * SSCALE; uh[i] = p[KBIN + i] * SSCALE; }
  float mw = uw[0], mh = uh[0];
#pragma unroll
  for (int i = 1; i < KBIN; ++i) { mw = fmaxf(mw, uw[i]); mh = fmaxf(mh, uh[i]); }
  float ew[KBIN], eh[KBIN], sw = 0.f, sh = 0.f;
#pragma unroll
  for (int i = 0; i < KBIN; ++i) {
    ew[i] = __expf(uw[i] - mw); sw += ew[i];
    eh[i] = __expf(uh[i] - mh); sh += eh[i];
  }
  float fw = (1.f - MINW * KBIN) / sw, fh = (1.f - MINW * KBIN) / sh;

  // single pass: cumulative knots + predicated bin selection
  float cw = -TBND, ch = -TBND, cws = 0.f, chs = 0.f;
  float icw = -TBND, iw = 1.f, ich = -TBND, ih = 1.f;
  int idx = 0;
#pragma unroll
  for (int i = 0; i < KBIN; ++i) {
    cws += MINW + ew[i] * fw;
    chs += MINW + eh[i] * fh;
    float cwn = (i == KBIN - 1) ? TBND : (2.f * TBND * cws - TBND);
    float chn = (i == KBIN - 1) ? TBND : (2.f * TBND * chs - TBND);
    if (xc >= cw) { idx = i; icw = cw; iw = cwn - cw; ich = ch; ih = chn - ch; }
    cw = cwn; ch = chn;
  }
  float dk  = (idx == 0)        ? 1.f : MIND + softplusf(p[2 * KBIN + idx - 1]);
  float dk1 = (idx == KBIN - 1) ? 1.f : MIND + softplusf(p[2 * KBIN + idx]);
  float delta = ih / iw;
  float th  = (xc - icw) / iw;
  float om  = 1.f - th;
  float t1m = th * om;
  float den = delta + (dk + dk1 - 2.f * delta) * t1m;
  float y   = ich + ih * (delta * th * th + dk * t1m) / den;
  float dnum = delta * delta * (dk1 * th * th + 2.f * delta * t1m + dk * om * om);
  float ld = __logf(dnum) - 2.f * __logf(den);
  y  = inside ? y : x0;
  ld = inside ? ld : 0.f;
  Xdst[(size_t)b * F_DIM + f] = y;
#pragma unroll
  for (int o = 8; o; o >>= 1) ld += __shfl_xor(ld, o, 16);
  if (f == 0) ldout[b] += ld;
}

// ---------------- prep: mask-fused bf16 B-fragment packing ----------------
__device__ __forceinline__ float mmask(int mt, int n, int k) {
  switch (mt) {
    case 0:  return 1.f;                                        // Wc
    case 1:  return ((n % 15) + 1) >= (k + 1)        ? 1.f : 0.f; // m0
    case 2:  return ((n % 15) + 1) >= ((k % 15) + 1) ? 1.f : 0.f; // mh
    default: return ((n / 47) + 1) >  ((k % 15) + 1) ? 1.f : 0.f; // mf
  }
}

__global__ void pack_kernel(const float* __restrict__ W, __bf16* __restrict__ out,
                            int Ltot, int NT, int KC, int Kreal, int Nrows, int mt) {
  size_t t = (size_t)blockIdx.x * blockDim.x + threadIdx.x;
  size_t total = (size_t)Ltot * NT * KC * 512;
  if (t >= total) return;
  int hh = (int)(t & 15);
  int lane = (int)((t >> 4) & 31);
  size_t r = t >> 9;
  int kc = (int)(r % KC); r /= KC;
  int nt = (int)(r % NT); r /= NT;
  int lm = (int)r;
  int n = nt * 16 + (lane & 15);
  int k = kc * 32 + ((lane >> 4) << 3) + (hh & 7) + ((hh >> 3) << 4);   // ISA B layout: lanes>=16 -> K+8, halves 8..15 -> K+16
  float v = 0.f;
  if (k < Kreal) v = W[((size_t)lm * Nrows + n) * Kreal + k] * mmask(mt, n, k);
  out[t] = (__bf16)v;
}

__global__ void cvt_ctx_kernel(const float* __restrict__ c, __bf16* __restrict__ o, size_t n) {
  size_t t = (size_t)blockIdx.x * blockDim.x + threadIdx.x;
  if (t < n) o[t] = (__bf16)c[t];
}
__global__ void zero_kernel(float* __restrict__ p, size_t n) {
  size_t t = (size_t)blockIdx.x * blockDim.x + threadIdx.x;
  if (t < n) p[t] = 0.f;
}

// ---------------- host ----------------
extern "C" void kernel_launch(void* const* d_in, const int* in_sizes, int n_in,
                              void* d_out, int out_size, void* d_ws, size_t ws_size,
                              hipStream_t stream) {
  const float* x   = (const float*)d_in[0];
  const float* ctx = (const float*)d_in[1];
  const float* W0  = (const float*)d_in[2];
  const float* b0  = (const float*)d_in[3];
  const float* Wc  = (const float*)d_in[4];
  const float* bc  = (const float*)d_in[5];
  const float* Wr  = (const float*)d_in[6];
  const float* br  = (const float*)d_in[7];
  const float* Wf  = (const float*)d_in[8];
  const float* bfb = (const float*)d_in[9];
  const int B = in_sizes[0] / F_DIM;

  char* ws = (char*)d_ws;
  size_t off = 0;
  auto take = [&](size_t bytes) -> char* {
    char* p = ws + off;
    off = (off + bytes + 255) & ~(size_t)255;
    return p;
  };
  __bf16* ctxb = (__bf16*)take((size_t)B * CDIM * 2);
  __bf16* w0f  = (__bf16*)take((size_t)NLAYER * 16 * 1 * 512 * 2);
  __bf16* wcf  = (__bf16*)take((size_t)NLAYER * 16 * 8 * 512 * 2);
  __bf16* wrf  = (__bf16*)take((size_t)NLAYER * 4 * 16 * 8 * 512 * 2);
  __bf16* wff  = (__bf16*)take((size_t)NLAYER * 47 * 8 * 512 * 2);
  float* ping0 = (float*)take((size_t)B * F_DIM * 4);
  float* ping1 = (float*)take((size_t)B * F_DIM * 4);
  // size the spline-parameter chunk buffer to whatever workspace remains
  size_t rem = (ws_size > off) ? (ws_size - off) : 0;
  int chunk = (int)(rem / ((size_t)POUT * 4));
  chunk &= ~63;
  if (chunk > MAXCHUNK) chunk = MAXCHUNK;
  if (chunk < 64) chunk = 64;
  if (chunk > B) chunk = B;
  float* Pbuf = (float*)take((size_t)chunk * POUT * 4);

  float* xout  = (float*)d_out;
  float* ldout = xout + (size_t)B * F_DIM;

  { size_t n = (size_t)B;
    zero_kernel<<<(unsigned)((n + 255) / 256), 256, 0, stream>>>(ldout, n); }
  { size_t n = (size_t)B * CDIM;
    cvt_ctx_kernel<<<(unsigned)((n + 255) / 256), 256, 0, stream>>>(ctx, ctxb, n); }
  { size_t n = (size_t)NLAYER * 16 * 1 * 512;
    pack_kernel<<<(unsigned)((n + 255) / 256), 256, 0, stream>>>(W0, w0f, NLAYER, 16, 1, 16, HID, 1); }
  { size_t n = (size_t)NLAYER * 16 * 8 * 512;
    pack_kernel<<<(unsigned)((n + 255) / 256), 256, 0, stream>>>(Wc, wcf, NLAYER, 16, 8, 256, HID, 0); }
  { size_t n = (size_t)NLAYER * 4 * 16 * 8 * 512;
    pack_kernel<<<(unsigned)((n + 255) / 256), 256, 0, stream>>>(Wr, wrf, NLAYER * 4, 16, 8, 256, HID, 2); }
  { size_t n = (size_t)NLAYER * 47 * 8 * 512;
    pack_kernel<<<(unsigned)((n + 255) / 256), 256, 0, stream>>>(Wf, wff, NLAYER, 47, 8, 256, POUT, 3); }

  const size_t smem = (size_t)BT * HP * sizeof(float) + 2 * (size_t)BT * HBP * sizeof(__bf16);
  (void)hipFuncSetAttribute(reinterpret_cast<const void*>(made_kernel),
                            hipFuncAttributeMaxDynamicSharedMemorySize, (int)smem);

  float* pp[2] = {ping0, ping1};
  for (int rowbase = 0; rowbase < B; rowbase += chunk) {
    int rows = (B - rowbase < chunk) ? (B - rowbase) : chunk;
    for (int l = 0; l < NLAYER; ++l) {
      const float* xsrc = (l == 0) ? x : pp[l & 1];
      float* xdst = (l == NLAYER - 1) ? xout : pp[(l + 1) & 1];
      made_kernel<<<(unsigned)(rows / BT), 256, smem, stream>>>(
          xsrc, ctxb, w0f, wcf, wrf, wff, b0, bc, br, bfb, Pbuf, l, rowbase);
      spline_kernel<<<(unsigned)((rows * F_DIM + 255) / 256), 256, 0, stream>>>(
          xsrc, Pbuf, xdst, ldout, rowbase, rows);
    }
  }
}
